// AdaptiveGraphConv_75617194213762
// MI455X (gfx1250) — matrix operand order repaired and checked by reading, not compiled, not run on hardware
//
#include <hip/hip_runtime.h>
#include <hip/hip_bf16.h>

// Problem constants
#define BB 8
#define CC 256
#define TT 64
#define VV 468
#define VP 480          // V padded to 30 tiles of 16
#define NI 128          // INTER
#define NO 256          // OUT
#define NROWS 512       // theta(128)+phi(128)+g(256) rows fused
#define XST 264         // LDS row stride (halves) for X^T, 256 + 8 pad (bank skew)
#define BT (BB*TT)

typedef __attribute__((ext_vector_type(16))) _Float16 v16h;
typedef __attribute__((ext_vector_type(8)))  _Float16 v8h;
typedef __attribute__((ext_vector_type(8)))  float    v8f;

// ---------------- WMMA helpers (CDNA5 wave32, 16x16x32 f16 -> f32) ----------

__device__ __forceinline__ v8f wmma_f16(v16h a, v16h b, v8f c) {
#if defined(__gfx1250__)
  // (neg_a, A, neg_b, B, c_mod, C, reuse_a, reuse_b)
  return __builtin_amdgcn_wmma_f32_16x16x32_f16(false, a, false, b, (short)0, c,
                                                false, false);
#else
  (void)a; (void)b;
  return c;
#endif
}

// A fragment: matrix is row-major [m][k], p points at (m0,k0), stride in halves.
// Lane layout (ISA 16-bit A 16x32): half-wave h holds K = {8h..8h+7, 16+8h..16+8h+7}.
__device__ __forceinline__ v16h load_frag_a(const _Float16* p, int stride, int lane) {
  int m = lane & 15, h = lane >> 4;
  const _Float16* r = p + (size_t)m * stride + h * 8;
  v8h lo = *(const v8h*)(r);          // K = 8h .. 8h+7         (16B)
  v8h hi = *(const v8h*)(r + 16);     // K = 16+8h .. 16+8h+7   (16B)
  v16h a;
#pragma unroll
  for (int j = 0; j < 8; ++j) { a[j] = lo[j]; a[j + 8] = hi[j]; }
  return a;
}

// B fragment: B is (K x N); stored transposed row-major [n][k], p at (k0,n0).
// Lane layout (ISA 16-bit B): half-wave h holds K = {8h..8h+7, 16+8h..16+8h+7}.
__device__ __forceinline__ v16h load_frag_b(const _Float16* p, int stride, int lane) {
  int n = lane & 15, h = lane >> 4;
  const _Float16* r = p + (size_t)n * stride + h * 8;
  v8h lo = *(const v8h*)(r);          // 16B
  v8h hi = *(const v8h*)(r + 16);     // 16B
  v16h b;
#pragma unroll
  for (int j = 0; j < 8; ++j) { b[j] = lo[j]; b[j + 8] = hi[j]; }
  return b;
}

__device__ __forceinline__ void wave_lds_fence() {
#if defined(__gfx1250__)
  asm volatile("s_wait_dscnt 0x0" ::: "memory");   // cross-lane LDS visibility, same wave
#else
  __threadfence_block();
#endif
}

// ---------------- Kernel 1: pack weights to f16, biases, zero BN stats ------

__global__ void k_prep(const float* __restrict__ Wt, const float* __restrict__ bth,
                       const float* __restrict__ Wp, const float* __restrict__ bph,
                       const float* __restrict__ Wg, const float* __restrict__ bg,
                       _Float16* __restrict__ Wall, float* __restrict__ ball,
                       float* __restrict__ stats) {
  int i = blockIdx.x * blockDim.x + threadIdx.x;
  if (i < NROWS * CC) {
    int m = i / CC, c = i % CC;
    float w = (m < NI) ? Wt[m * CC + c]
            : (m < 2 * NI) ? Wp[(m - NI) * CC + c]
            : Wg[(m - 2 * NI) * CC + c];
    Wall[i] = (_Float16)w;
  }
  if (i < NROWS)
    ball[i] = (i < NI) ? bth[i] : (i < 2 * NI) ? bph[i - NI] : bg[i - 2 * NI];
  if (i < 2 * NO) stats[i] = 0.f;   // gsum[256] ++ gsumsq[256]
}

// ---------------- Kernel 2: fused projections  Y = Wall @ X  per (b,t) ------
// LDS: X^T[VP][XST] f16 (~248KB) + bias copy. Output layouts:
//   ThetaT/PhiT: [bt][v(480)][i(128)]  (row-major in i)  -> attn A/B frags
//   G:           [bt][o(256)][v(480)]  (row-major in v)  -> attn B frags

__global__ void k_proj(const float* __restrict__ x,
                       const _Float16* __restrict__ Wall,
                       const float* __restrict__ ball,
                       _Float16* __restrict__ ThetaT,
                       _Float16* __restrict__ PhiT,
                       _Float16* __restrict__ G) {
  extern __shared__ char smem[];
  _Float16* Xt  = (_Float16*)smem;                       // [VP][XST]
  float*    bsh = (float*)(smem + (size_t)VP * XST * 2); // [NROWS]

  const int bt = blockIdx.x, b = bt / TT, t = bt % TT;
  const int tid = threadIdx.x, wid = tid >> 5, lane = tid & 31;

  for (int i = tid; i < NROWS; i += 256) bsh[i] = ball[i];
  // zero-pad rows v = VV..VP-1 (so padded theta/phi/g stay finite)
  for (int i = tid; i < (VP - VV) * CC; i += 256) {
    int v = VV + i / CC, c = i % CC;
    Xt[(size_t)v * XST + c] = (_Float16)0.f;
  }
  // stage X^T: x[b,c,t,v] -> Xt[v][c], coalesced in v
  const float* xb = x + (size_t)b * CC * TT * VV + (size_t)t * VV;
  for (int c = 0; c < CC; ++c) {
    const float* xr = xb + (size_t)c * TT * VV;
    for (int v = tid; v < VV; v += 256)
      Xt[(size_t)v * XST + c] = (_Float16)xr[v];
  }
  __syncthreads();

  const int half = lane >> 4, n = lane & 15;
  // each wave owns m-tiles wid, wid+8, ... (A fragments preloaded => Wall read once/WG)
  for (int mt = wid; mt < NROWS / 16; mt += 8) {
    const int m0 = mt * 16;
    v16h afrag[8];
#pragma unroll
    for (int kk = 0; kk < 8; ++kk)
      afrag[kk] = load_frag_a(Wall + (size_t)m0 * CC + kk * 32, CC, lane);
    float bias8[8];
#pragma unroll
    for (int r = 0; r < 8; ++r) bias8[r] = bsh[m0 + 8 * half + r];

    for (int nt = 0; nt < VP / 16; ++nt) {
      const int v = nt * 16 + n;
      v8f acc = {};
#pragma unroll
      for (int kk = 0; kk < 8; ++kk) {
        v16h bfrag = load_frag_b(Xt + (size_t)(nt * 16) * XST + kk * 32, XST, lane);
        acc = wmma_f16(afrag[kk], bfrag, acc);
      }
      // D: acc[r] = (row m0+8*half+r, col v)
      if (m0 < 2 * NI) {   // theta or phi -> transposed [v][i], contiguous v8h store
        v8h hv;
#pragma unroll
        for (int r = 0; r < 8; ++r) hv[r] = (_Float16)(acc[r] + bias8[r]);
        _Float16* base = (m0 < NI) ? ThetaT : PhiT;
        int i0 = (m0 < NI ? m0 : m0 - NI) + 8 * half;
        *(v8h*)(base + ((size_t)bt * VP + v) * NI + i0) = hv;
      } else {             // g -> [o][v]
#pragma unroll
        for (int r = 0; r < 8; ++r) {
          int o = (m0 - 2 * NI) + 8 * half + r;
          G[((size_t)bt * NO + o) * VP + v] = (_Float16)(acc[r] + bias8[r]);
        }
      }
    }
  }
}

// ---------------- Kernel 3: attention + output + BN partial sums ------------
// One WG per (b,t); each wave owns PAIRS of 16-row v-tiles so every G tile
// fetched from L2 feeds two accumulator sets (halves G traffic).

__global__ void k_attn(const _Float16* __restrict__ ThetaT,
                       const _Float16* __restrict__ PhiT,
                       const _Float16* __restrict__ G,
                       float* __restrict__ out,
                       float* __restrict__ gsum, float* __restrict__ gsumsq) {
  extern __shared__ char smem[];
  _Float16* Sall   = (_Float16*)smem;                          // 8 waves * 2*16*VP f16
  float*    lsum   = (float*)(smem + (size_t)8 * 2 * 16 * VP * 2); // [NO]
  float*    lsumsq = lsum + NO;

  const int bt = blockIdx.x, b = bt / TT, t = bt % TT;
  const int tid = threadIdx.x, wid = tid >> 5, lane = tid & 31;
  const int half = lane >> 4, n = lane & 15;

  for (int i = tid; i < 2 * NO; i += 256) lsum[i] = 0.f;
  __syncthreads();

  const _Float16* Th = ThetaT + (size_t)bt * VP * NI;
  const _Float16* Ph = PhiT   + (size_t)bt * VP * NI;
  const _Float16* Gm = G      + (size_t)bt * NO * VP;
  const float scale = 0.08838834764831845f;  // 1/sqrt(128)

  _Float16* Sw = Sall + (size_t)wid * 2 * 16 * VP;   // this wave's 2 prob tiles

  for (int p = wid; p < (VP / 16) / 2; p += 8) {     // 15 tile-pairs
#pragma unroll
    for (int which = 0; which < 2; ++which) {
      const int v0 = (2 * p + which) * 16;
      _Float16* S = Sw + which * 16 * VP;
      // ---- scores: Theta^T(v-tile) x Phi, K = 128 ----
      v16h afrag[4];
#pragma unroll
      for (int kk = 0; kk < 4; ++kk)
        afrag[kk] = load_frag_a(Th + (size_t)v0 * NI + kk * 32, NI, lane);
      for (int wt = 0; wt < VP / 16; ++wt) {
        v8f acc = {};
#pragma unroll
        for (int kk = 0; kk < 4; ++kk) {
          v16h bfrag = load_frag_b(Ph + (size_t)(wt * 16) * NI + kk * 32, NI, lane);
          acc = wmma_f16(afrag[kk], bfrag, acc);
        }
#pragma unroll
        for (int r = 0; r < 8; ++r)
          S[(size_t)(r + 8 * half) * VP + wt * 16 + n] = (_Float16)(acc[r] * scale);
      }
      wave_lds_fence();
      // ---- softmax per row; 2 lanes/row (half 0: w<240, half 1: 240..467) ----
      {
        const int row = n;
        const int wlo = half * 240, whi = half ? VV : 240;
        float mx = -3.0e38f;
        for (int w = wlo; w < whi; ++w)
          mx = fmaxf(mx, (float)S[(size_t)row * VP + w]);
        mx = fmaxf(mx, __shfl_xor(mx, 16));
        float sum = 0.f;
        for (int w = wlo; w < whi; ++w) {
          float e = __expf((float)S[(size_t)row * VP + w] - mx);
          sum += e;
          S[(size_t)row * VP + w] = (_Float16)e;
        }
        sum += __shfl_xor(sum, 16);
        float inv = 1.f / sum;
        for (int w = wlo; w < whi; ++w)
          S[(size_t)row * VP + w] =
              (_Float16)((float)S[(size_t)row * VP + w] * inv);
        if (half)   // zero padded columns so padded-G rows contribute nothing
          for (int w = VV; w < VP; ++w) S[(size_t)row * VP + w] = (_Float16)0.f;
      }
    }
    wave_lds_fence();
    // ---- out = P @ G for both row-tiles, sharing every G fragment ----
    const int v0a = (2 * p) * 16, v0b = (2 * p + 1) * 16;
    for (int ot = 0; ot < NO / 16; ++ot) {
      v8f acc0 = {}, acc1 = {};
      for (int k0 = 0; k0 < VP; k0 += 32) {
        v16h bfrag = load_frag_b(Gm + (size_t)(ot * 16) * VP + k0, VP, lane);
        v16h a0 = load_frag_a(Sw + k0, VP, lane);
        v16h a1 = load_frag_a(Sw + 16 * VP + k0, VP, lane);
        acc0 = wmma_f16(a0, bfrag, acc0);
        acc1 = wmma_f16(a1, bfrag, acc1);
      }
      const int o = ot * 16 + n;
      float* orow = out + (((size_t)b * NO + o) * TT + t) * VV;
      float s8 = 0.f, q8 = 0.f;
#pragma unroll
      for (int r = 0; r < 8; ++r) {
        int v = v0a + 8 * half + r;
        if (v < VV) { orow[v] = acc0[r]; s8 += acc0[r]; q8 += acc0[r] * acc0[r]; }
      }
#pragma unroll
      for (int r = 0; r < 8; ++r) {
        int v = v0b + 8 * half + r;
        if (v < VV) { orow[v] = acc1[r]; s8 += acc1[r]; q8 += acc1[r] * acc1[r]; }
      }
      atomicAdd(&lsum[o], s8);     // ds_add_f32
      atomicAdd(&lsumsq[o], q8);
    }
  }
  __syncthreads();
  for (int o = tid; o < NO; o += 256) {
    atomicAdd(&gsum[o], lsum[o]);
    atomicAdd(&gsumsq[o], lsumsq[o]);
  }
}

// ---------------- Kernel 4: finalize BN scale/shift per channel -------------

__global__ void k_bnstats(const float* __restrict__ gsum,
                          const float* __restrict__ gsumsq,
                          const float* __restrict__ gamma,
                          const float* __restrict__ beta,
                          float* __restrict__ bnab) {
  int o = blockIdx.x * blockDim.x + threadIdx.x;
  if (o < NO) {
    const float Ninv = 1.0f / (float)(BB * TT * VV);
    float mean = gsum[o] * Ninv;
    float var  = gsumsq[o] * Ninv - mean * mean;
    float sc   = gamma[o] * rsqrtf(var + 1e-5f);
    bnab[2 * o]     = sc;
    bnab[2 * o + 1] = beta[o] - mean * sc;
  }
}

// ---------------- Kernel 5: apply BN in place -------------------------------

__global__ void k_bnapply(float* __restrict__ out, const float* __restrict__ bnab) {
  const unsigned ntot = (unsigned)BB * NO * TT * VV;
  unsigned i = blockIdx.x * blockDim.x + threadIdx.x;
  const unsigned stride = gridDim.x * blockDim.x;
  for (; i < ntot; i += stride) {
    int o = (int)((i / (TT * VV)) % NO);
    out[i] = out[i] * bnab[2 * o] + bnab[2 * o + 1];
  }
}

// ---------------- Launch ----------------------------------------------------

extern "C" void kernel_launch(void* const* d_in, const int* in_sizes, int n_in,
                              void* d_out, int out_size, void* d_ws, size_t ws_size,
                              hipStream_t stream) {
  (void)in_sizes; (void)n_in; (void)out_size; (void)ws_size;
  const float* x   = (const float*)d_in[0];
  const float* Wt  = (const float*)d_in[1];
  const float* bth = (const float*)d_in[2];
  const float* Wp  = (const float*)d_in[3];
  const float* bph = (const float*)d_in[4];
  const float* Wg  = (const float*)d_in[5];
  const float* bg  = (const float*)d_in[6];
  const float* gam = (const float*)d_in[7];
  const float* bet = (const float*)d_in[8];
  float* out = (float*)d_out;

  char* ws = (char*)d_ws;
  size_t off = 0;
  auto take = [&](size_t bytes) { char* p = ws + off; off += (bytes + 255) & ~(size_t)255; return p; };
  _Float16* Wall   = (_Float16*)take((size_t)NROWS * CC * 2);        // 256 KB
  float*    ball   = (float*)   take((size_t)NROWS * 4);
  float*    stats  = (float*)   take((size_t)2 * NO * 4);            // gsum|gsumsq
  float*    bnab   = (float*)   take((size_t)2 * NO * 4);
  _Float16* ThetaT = (_Float16*)take((size_t)BT * VP * NI * 2);      // 60 MB
  _Float16* PhiT   = (_Float16*)take((size_t)BT * VP * NI * 2);      // 60 MB
  _Float16* G      = (_Float16*)take((size_t)BT * NO * VP * 2);      // 120 MB
  float* gsum = stats, *gsumsq = stats + NO;

  k_prep<<<(NROWS * CC + 255) / 256, 256, 0, stream>>>(Wt, bth, Wp, bph, Wg, bg,
                                                       Wall, ball, stats);

  const size_t smem_proj = (size_t)VP * XST * 2 + (size_t)NROWS * 4;   // ~250 KB
  k_proj<<<BT, 256, smem_proj, stream>>>(x, Wall, ball, ThetaT, PhiT, G);

  const size_t smem_attn = (size_t)8 * 2 * 16 * VP * 2 + (size_t)2 * NO * 4; // ~242 KB
  k_attn<<<BT, 256, smem_attn, stream>>>(ThetaT, PhiT, G, out, gsum, gsumsq);

  k_bnstats<<<1, 256, 0, stream>>>(gsum, gsumsq, gam, bet, bnab);
  k_bnapply<<<8192, 256, 0, stream>>>(out, bnab);
}
